// SequenceDropoutAttention_88811333747327
// MI455X (gfx1250) — matrix operand order repaired and checked by reading, not compiled, not run on hardware
//
#include <hip/hip_runtime.h>
#include <hip/hip_bf16.h>

// ---- problem dims (from reference) ----
#define BB 2
#define NN 2048
#define CC 1024
#define HH 16
#define DD 64
#define HD 1024          // H*D
#define QKVO 3072        // 3*H*D
#define MTOK 4096        // B*N
#define KKEEP 1228
#define KPAD 1248        // multiple of 32
#define KBLKS (KPAD / 32)

typedef __bf16 v16bf __attribute__((ext_vector_type(16)));
typedef __bf16 v8bf  __attribute__((ext_vector_type(8)));
typedef float  v8f   __attribute__((ext_vector_type(8)));

static __device__ __forceinline__ __bf16 f2bf(float f) {
  unsigned u = __builtin_bit_cast(unsigned, f);
  unsigned r = u + 0x7FFFu + ((u >> 16) & 1u);     // RNE
  unsigned short hs = (unsigned short)(r >> 16);
  return __builtin_bit_cast(__bf16, hs);
}

static __device__ __forceinline__ void wait_async0() {
#if __has_builtin(__builtin_amdgcn_s_wait_asynccnt)
  __builtin_amdgcn_s_wait_asynccnt(0);
#else
  asm volatile("s_wait_asynccnt 0x0" ::: "memory");
#endif
}

// Async global->LDS 16-byte copy (gfx1250, ASYNCcnt-tracked).
// GVS addressing: mem = SGPR64 base + VGPR32 offset; dst = per-lane LDS byte addr.
static __device__ __forceinline__ void async_ld_b128(unsigned lds_addr, int voff, const void* sbase) {
  asm volatile("global_load_async_to_lds_b128 %0, %1, %2"
               :: "v"(lds_addr), "v"(voff), "s"(sbase)
               : "memory");
}

// ---------------- f32 -> bf16 convert ----------------
__global__ void cvt_f32_bf16(const float* __restrict__ in, __bf16* __restrict__ out, int n) {
  int i = blockIdx.x * blockDim.x + threadIdx.x;
  if (i < n) out[i] = f2bf(in[i]);
}

// ---------------- exact stable top-K per (b,h) ----------------
__global__ __launch_bounds__(256) void topk_kernel(const int* __restrict__ seq_mask,
                                                   const unsigned char* __restrict__ mask,
                                                   int* __restrict__ idx_out) {
  const int bh = blockIdx.x;            // B*H blocks
  const int b = bh / HH;
  const int tid = threadIdx.x;
  const int* sm = seq_mask + (size_t)bh * NN;
  const unsigned char* mb = mask + (size_t)b * NN;

  __shared__ int hist[11];
  __shared__ int tg[256], te[256];
  __shared__ int s_thr, s_m;

  if (tid < 11) hist[tid] = 0;
  __syncthreads();

  int lv[8];
  const int base = tid * 8;             // contiguous chunk preserves index order
  for (int i = 0; i < 8; ++i) {
    int v = mb[base + i] ? -1 : sm[base + i];
    lv[i] = v;
    atomicAdd(&hist[v + 1], 1);
  }
  __syncthreads();

  if (tid == 0) {
    int g = 0, t = -1;
    for (int v = 9; v >= -1; --v) {
      int c = hist[v + 1];
      if (g + c >= KKEEP) { t = v; break; }
      g += c;
    }
    s_thr = t;
    s_m = KKEEP - g;                    // how many ==t survive (earliest indices)
  }
  __syncthreads();
  const int t = s_thr, m = s_m;

  int gc = 0, ec = 0;
  for (int i = 0; i < 8; ++i) { gc += (lv[i] > t); ec += (lv[i] == t); }
  tg[tid] = gc; te[tid] = ec;
  __syncthreads();
  if (tid == 0) {
    int ag = 0, ae = 0;
    for (int j = 0; j < 256; ++j) {
      int g = tg[j]; tg[j] = ag; ag += g;
      int e = te[j]; te[j] = ae; ae += e;
    }
  }
  __syncthreads();

  int gp = tg[tid], ep = te[tid];
  int* outp = idx_out + (size_t)bh * KPAD;
  for (int i = 0; i < 8; ++i) {
    int v = lv[i];
    if (v > t) {
      int em = ep < m ? ep : m;
      outp[gp + em] = base + i;
      ++gp;
    } else if (v == t) {
      if (ep < m) outp[gp + ep] = base + i;
      ++ep;
    }
  }
}

// ---------------- QKV GEMM: [4096,1024] x [3072,1024]^T, bf16 WMMA ----------------
__global__ __launch_bounds__(256) void qkv_gemm(const __bf16* __restrict__ A,
                                                const __bf16* __restrict__ W,
                                                __bf16* __restrict__ Qb,
                                                __bf16* __restrict__ Kf,
                                                __bf16* __restrict__ Vf) {
  const int lane = threadIdx.x & 31;
  const int wave = threadIdx.x >> 5;
  const bool hi  = lane >= 16;
  const int lrow = lane & 15;
  const int m0 = (blockIdx.y * 8 + wave) * 16;
  const int n0 = blockIdx.x * 64;

  v8f acc[4];
  for (int tt = 0; tt < 4; ++tt) for (int r = 0; r < 8; ++r) acc[tt][r] = 0.f;

  const __bf16* arow = A + (size_t)(m0 + lrow) * CC;
  for (int ks = 0; ks < CC; ks += 32) {
    v16bf a;
    {
      int c0 = ks + (hi ? 16 : 0);      // A chunks: contiguous K, k = 16h + s (same as B)
      v8bf lo = *(const v8bf*)(arow + c0);
      v8bf h2 = *(const v8bf*)(arow + c0 + 8);
      for (int i = 0; i < 8; ++i) { a[i] = lo[i]; a[i + 8] = h2[i]; }
    }
    for (int tt = 0; tt < 4; ++tt) {
      int col = n0 + tt * 16 + lrow;
      int db  = ks + (hi ? 16 : 0);     // B layout: per-lane column, contiguous K
      v16bf bm = *(const v16bf*)(W + (size_t)col * CC + db);
      acc[tt] = __builtin_amdgcn_wmma_f32_16x16x32_bf16(false, a, false, bm,
                                                        (short)0, acc[tt], false, false);
    }
  }
  for (int tt = 0; tt < 4; ++tt) {
    int col = n0 + tt * 16 + lrow;
    int h = col / 192, rem = col % 192, d = rem / 3, e = rem % 3;
    for (int r = 0; r < 8; ++r) {
      int mrow = m0 + r + (hi ? 8 : 0);
      int b = mrow / NN, n = mrow % NN;
      size_t di = ((size_t)(b * HH + h) * NN + n) * DD + d;
      __bf16 v = f2bf(acc[tt][r]);
      if (e == 0) Qb[di] = v; else if (e == 1) Kf[di] = v; else Vf[di] = v;
    }
  }
}

// ---------------- gather kept K rows (compact) and V (transposed) ----------------
__global__ void gather_kv(const __bf16* __restrict__ Kf, const __bf16* __restrict__ Vf,
                          const int* __restrict__ idx,
                          __bf16* __restrict__ Kc, __bf16* __restrict__ Vt) {
  int g = blockIdx.x * blockDim.x + threadIdx.x;
  if (g >= BB * HH * KPAD) return;
  int bh = g / KPAD, j = g % KPAD;
  __bf16* kc = Kc + ((size_t)bh * KPAD + j) * DD;
  if (j < KKEEP) {
    int src = idx[(size_t)bh * KPAD + j];
    const __bf16* kr = Kf + ((size_t)bh * NN + src) * DD;
    const __bf16* vr = Vf + ((size_t)bh * NN + src) * DD;
    for (int d = 0; d < DD; ++d) {
      kc[d] = kr[d];
      Vt[((size_t)bh * DD + d) * KPAD + j] = vr[d];
    }
  } else {
    __bf16 z = f2bf(0.f);
    for (int d = 0; d < DD; ++d) {
      kc[d] = z;
      Vt[((size_t)bh * DD + d) * KPAD + j] = z;
    }
  }
}

// ---------------- flash attention over kept keys, bf16 WMMA ----------------
// grid: (B*H, N/64), block 128 (4 waves, one 16-query tile each).
// All 4 waves share (b,h): K/V 32-key tiles are staged once per block into LDS
// with double-buffered GLOBAL_LOAD_ASYNC_TO_LDS_B128 (ASYNCcnt), overlapping the
// next block's staging with the current block's WMMA + softmax.
__global__ __launch_bounds__(128) void attn_kernel(const __bf16* __restrict__ Qb,
                                                   const __bf16* __restrict__ Kc,
                                                   const __bf16* __restrict__ Vt,
                                                   const int* __restrict__ idx,
                                                   const float* __restrict__ pos_bias,
                                                   const unsigned char* __restrict__ mask,
                                                   __bf16* __restrict__ Oa) {
  __shared__ __align__(16) __bf16 Kt[2][32 * 64];    // [j][d], 4 KB per buffer
  __shared__ __align__(16) __bf16 Vts[2][64 * 32];   // [d][j], 4 KB per buffer
  __shared__ __align__(16) __bf16 Pls[4 * 16 * 32];  // per-wave P tile relay
  const int tid  = threadIdx.x;
  const int lane = threadIdx.x & 31;
  const int wave = threadIdx.x >> 5;
  const bool hi  = lane >= 16;
  const int lrow = lane & 15;
  const int bh = blockIdx.x;
  const int b = bh / HH, h = bh % HH;
  const int i0 = (blockIdx.y * 4 + wave) * 16;

  // ---- per-thread async-staging chunk descriptors (16B chunks) ----
  // K tile: 4 KB contiguous -> chunk q = tid + it*128, byte off q*16
  // V tile: row d = q>>2 (stride KPAD*2 B), col chunk c = (q&3)*8 elements
  unsigned kLds[2][2], vLds[2][2];
  int kGo[2], vGo[2];
  for (int it = 0; it < 2; ++it) {
    int q = tid + it * 128;
    kGo[it] = q * 16;
    int d = q >> 2, c = (q & 3) * 8;
    vGo[it] = 2 * ((bh * DD + d) * KPAD + c);
    for (int buf = 0; buf < 2; ++buf) {
      kLds[buf][it] = (unsigned)(uintptr_t)(&Kt[buf][0]) + (unsigned)(q * 16);
      vLds[buf][it] = (unsigned)(uintptr_t)(&Vts[buf][0]) + (unsigned)((d * 32 + c) * 2);
    }
  }
  auto stage = [&](int j0, int buf) {
    const char* kb = (const char*)Kc + 2 * (((size_t)bh * KPAD + j0) * DD);
    const char* vb = (const char*)Vt + 2 * (size_t)j0;
    for (int it = 0; it < 2; ++it) {
      async_ld_b128(kLds[buf][it], kGo[it], kb);
      async_ld_b128(vLds[buf][it], vGo[it], vb);
    }
  };

  // Q tile in A layout, two K-steps of 32 covering D=64
  v16bf aq[2];
  {
    const __bf16* qrow = Qb + ((size_t)bh * NN + i0 + lrow) * DD;
    for (int s = 0; s < 2; ++s) {
      int c0 = s * 32 + (hi ? 16 : 0);
      v8bf lo = *(const v8bf*)(qrow + c0);
      v8bf h2 = *(const v8bf*)(qrow + c0 + 8);
      for (int i = 0; i < 8; ++i) { aq[s][i] = lo[i]; aq[s][i + 8] = h2[i]; }
    }
  }

  float mrow[8], lsum[8];
  v8f o[4];
  for (int r = 0; r < 8; ++r) { mrow[r] = -1e30f; lsum[r] = 0.f; }
  for (int tt = 0; tt < 4; ++tt) for (int r = 0; r < 8; ++r) o[tt][r] = 0.f;

  const float* pb = pos_bias + (size_t)h * NN * NN;
  const unsigned char* mb = mask + (size_t)b * NN;
  int qi[8], qm[8];
  for (int r = 0; r < 8; ++r) { qi[r] = i0 + r + (hi ? 8 : 0); qm[r] = mb[qi[r]]; }

  __bf16* pw = Pls + wave * 16 * 32;

  stage(0, 0);                              // prologue stage of block 0
  for (int blk = 0; blk < KBLKS; ++blk) {
    const int j0 = blk * 32;
    const int buf = blk & 1;
    wait_async0();                          // own async chunks for `buf` done
    __syncthreads();                        // all waves' chunks visible; prev buf free
    if (blk + 1 < KBLKS) stage(j0 + 32, buf ^ 1);

    const __bf16* KtB = &Kt[buf][0];
    const __bf16* VtB = &Vts[buf][0];

    v8f sc[2];
    for (int tt = 0; tt < 2; ++tt) {
      int jslot = j0 + tt * 16 + lrow;
      int valid = jslot < KKEEP;
      int kj = valid ? idx[(size_t)bh * KPAD + jslot] : 0;
      int km = valid ? (int)mb[kj] : 1;
      v8f s; for (int r = 0; r < 8; ++r) s[r] = 0.f;
      for (int st = 0; st < 2; ++st) {
        int db = st * 32 + (hi ? 16 : 0);
        v16bf bk = *(const v16bf*)(KtB + (tt * 16 + lrow) * DD + db);   // ds_load
        s = __builtin_amdgcn_wmma_f32_16x16x32_bf16(false, aq[st], false, bk,
                                                    (short)0, s, false, false);
      }
      for (int r = 0; r < 8; ++r) {
        float sv = s[r] * 0.125f;                          // D^-0.5
        if (valid) sv += pb[(size_t)qi[r] * NN + kj];      // pos-bias gather (HBM bound)
        bool msk = !valid || (kj > qi[r]) || (qm[r] && km);
        s[r] = msk ? -1e30f : sv;
      }
      sc[tt] = s;
    }
    // online softmax: rows live per-VGPR within each 16-lane half -> width-16 reductions
    float alpha[8];
    for (int r = 0; r < 8; ++r) {
      float v = fmaxf(sc[0][r], sc[1][r]);
      for (int ofs = 1; ofs < 16; ofs <<= 1) v = fmaxf(v, __shfl_xor(v, ofs, 16));
      float mn = fmaxf(mrow[r], v);
      alpha[r] = __expf(mrow[r] - mn);
      mrow[r] = mn;
    }
    for (int tt = 0; tt < 2; ++tt) {
      for (int r = 0; r < 8; ++r) {
        float s = sc[tt][r];
        float p = (s <= -1e29f) ? 0.f : __expf(s - mrow[r]);
        sc[tt][r] = p;
        pw[(r + (hi ? 8 : 0)) * 32 + tt * 16 + lrow] = f2bf(p);  // C-layout -> LDS
      }
    }
    for (int r = 0; r < 8; ++r) {
      float v = sc[0][r] + sc[1][r];
      for (int ofs = 1; ofs < 16; ofs <<= 1) v += __shfl_xor(v, ofs, 16);
      lsum[r] = lsum[r] * alpha[r] + v;
      for (int tt = 0; tt < 4; ++tt) o[tt][r] *= alpha[r];
    }
    // reload P in A layout (wave-internal DS ops are in-order)
    v16bf ap;
    {
      const __bf16* prow = pw + lrow * 32;
      int c0 = hi ? 16 : 0;
      v8bf lo = *(const v8bf*)(prow + c0);
      v8bf h2 = *(const v8bf*)(prow + c0 + 8);
      for (int i = 0; i < 8; ++i) { ap[i] = lo[i]; ap[i + 8] = h2[i]; }
    }
    for (int tt = 0; tt < 4; ++tt) {
      v16bf bv = *(const v16bf*)(VtB + (tt * 16 + lrow) * 32 + (hi ? 16 : 0)); // ds_load
      o[tt] = __builtin_amdgcn_wmma_f32_16x16x32_bf16(false, ap, false, bv,
                                                      (short)0, o[tt], false, false);
    }
    __syncthreads();                        // all waves done reading `buf` this iter
  }
  for (int tt = 0; tt < 4; ++tt) {
    for (int r = 0; r < 8; ++r) {
      float denom = fmaxf(lsum[r], 1e-30f);
      Oa[((size_t)(b * NN + qi[r])) * HD + h * DD + tt * 16 + lrow] = f2bf(o[tt][r] / denom);
    }
  }
}

// ---------------- output GEMM: [4096,1024] x [1024,1024]^T -> f32 ----------------
__global__ __launch_bounds__(256) void out_gemm(const __bf16* __restrict__ A,
                                                const __bf16* __restrict__ W,
                                                float* __restrict__ Y) {
  const int lane = threadIdx.x & 31;
  const int wave = threadIdx.x >> 5;
  const bool hi  = lane >= 16;
  const int lrow = lane & 15;
  const int m0 = (blockIdx.y * 8 + wave) * 16;
  const int n0 = blockIdx.x * 64;

  v8f acc[4];
  for (int tt = 0; tt < 4; ++tt) for (int r = 0; r < 8; ++r) acc[tt][r] = 0.f;

  const __bf16* arow = A + (size_t)(m0 + lrow) * HD;
  for (int ks = 0; ks < HD; ks += 32) {
    v16bf a;
    {
      int c0 = ks + (hi ? 16 : 0);
      v8bf lo = *(const v8bf*)(arow + c0);
      v8bf h2 = *(const v8bf*)(arow + c0 + 8);
      for (int i = 0; i < 8; ++i) { a[i] = lo[i]; a[i + 8] = h2[i]; }
    }
    for (int tt = 0; tt < 4; ++tt) {
      int col = n0 + tt * 16 + lrow;
      int db  = ks + (hi ? 16 : 0);
      v16bf bm = *(const v16bf*)(W + (size_t)col * HD + db);
      acc[tt] = __builtin_amdgcn_wmma_f32_16x16x32_bf16(false, a, false, bm,
                                                        (short)0, acc[tt], false, false);
    }
  }
  for (int tt = 0; tt < 4; ++tt) {
    for (int r = 0; r < 8; ++r) {
      int mrow = m0 + r + (hi ? 8 : 0);
      Y[(size_t)mrow * CC + n0 + tt * 16 + lrow] = acc[tt][r];
    }
  }
}

// ---------------- host launcher ----------------
extern "C" void kernel_launch(void* const* d_in, const int* in_sizes, int n_in,
                              void* d_out, int out_size, void* d_ws, size_t ws_size,
                              hipStream_t stream) {
  (void)in_sizes; (void)n_in; (void)out_size; (void)ws_size;
  const float* x        = (const float*)d_in[0];
  const float* Wqkv     = (const float*)d_in[1];
  const float* Wout     = (const float*)d_in[2];
  const float* pos_bias = (const float*)d_in[3];
  const int*   seq_mask = (const int*)d_in[4];
  const unsigned char* mask = (const unsigned char*)d_in[5];
  float* out = (float*)d_out;

  char* ws = (char*)d_ws;
  size_t off = 0;
  auto alloc = [&](size_t bytes) -> void* {
    void* p = ws + off;
    off += (bytes + 255) & ~(size_t)255;
    return p;
  };
  __bf16* x_bf  = (__bf16*)alloc((size_t)MTOK * CC * 2);
  __bf16* Wq_bf = (__bf16*)alloc((size_t)QKVO * CC * 2);
  __bf16* Wo_bf = (__bf16*)alloc((size_t)CC * HD * 2);
  __bf16* Qb    = (__bf16*)alloc((size_t)BB * HH * NN * DD * 2);
  __bf16* Kf    = (__bf16*)alloc((size_t)BB * HH * NN * DD * 2);
  __bf16* Vf    = (__bf16*)alloc((size_t)BB * HH * NN * DD * 2);
  __bf16* Kc    = (__bf16*)alloc((size_t)BB * HH * KPAD * DD * 2);
  __bf16* Vt    = (__bf16*)alloc((size_t)BB * HH * DD * KPAD * 2);
  int*    idx   = (int*)alloc((size_t)BB * HH * KPAD * 4);
  __bf16* Oa    = (__bf16*)alloc((size_t)MTOK * HD * 2);

  const int nx = MTOK * CC, nwq = QKVO * CC, nwo = CC * HD;
  cvt_f32_bf16<<<(nx  + 255) / 256, 256, 0, stream>>>(x,    x_bf,  nx);
  cvt_f32_bf16<<<(nwq + 255) / 256, 256, 0, stream>>>(Wqkv, Wq_bf, nwq);
  cvt_f32_bf16<<<(nwo + 255) / 256, 256, 0, stream>>>(Wout, Wo_bf, nwo);

  topk_kernel<<<BB * HH, 256, 0, stream>>>(seq_mask, mask, idx);

  qkv_gemm<<<dim3(QKVO / 64, MTOK / 128), 256, 0, stream>>>(x_bf, Wq_bf, Qb, Kf, Vf);

  gather_kv<<<(BB * HH * KPAD + 255) / 256, 256, 0, stream>>>(Kf, Vf, idx, Kc, Vt);

  attn_kernel<<<dim3(BB * HH, NN / 64), 128, 0, stream>>>(Qb, Kc, Vt, idx, pos_bias, mask, Oa);

  out_gemm<<<dim3(CC / 64, MTOK / 128), 256, 0, stream>>>(Oa, Wo_bf, out);
}
